// GATNet_18786186952892
// MI455X (gfx1250) — matrix operand order repaired and checked by reading, not compiled, not run on hardware
//
#include <hip/hip_runtime.h>
#include <hip/hip_bf16.h>

// ---------------------------------------------------------------------------
// GATNet (ParticleNet-style) forward for gfx1250 / MI455X.
// B=256 jets, N=128 nodes, K=16 knn, F_IN=34, HID=70, HEADS=4, DM=280, NCLS=5
//
// GEMMs: v_wmma_f32_16x16x32_bf16 on zero-padded bf16 global shadows.
// Staging: double-buffered LDS fed by gfx1250 async global->LDS DMA
// (GLOBAL_LOAD_ASYNC_TO_LDS_B128, ASYNCcnt), overlapping tile k+1 DMA with
// the WMMAs on tile k.
// ---------------------------------------------------------------------------

#define BATCH   256
#define NNODES  128
#define KNN     16
#define F_IN    34
#define HID     70
#define HEADS   4
#define DM      280                 // HID*HEADS
#define NTOT    (BATCH * NNODES)    // 32768
#define NLAYERS 9
#define KPAD_DM 288                 // DM padded to mult of 32
#define KPAD_F  64                  // F_IN padded to mult of 32
#define NPAD_DM 320                 // DM padded to mult of 64
#define NPAD_H  128                 // HID padded to mult of 64

typedef __attribute__((ext_vector_type(16))) __bf16 v16bf;
typedef __attribute__((ext_vector_type(8)))  float  v8f;
typedef int v4i __attribute__((vector_size(16)));   // matches builtin param type

struct B256 { uint4 lo; uint4 hi; };   // 32 bytes == v16bf

// ---- gfx1250 async global->LDS path (guarded: falls back to sync staging) --
#if __has_builtin(__builtin_amdgcn_global_load_async_to_lds_b128)
#define HAVE_ASYNC_LDS 1
#define ASYNC_B128(dst, src)                                                   \
  __builtin_amdgcn_global_load_async_to_lds_b128(                              \
      (__attribute__((address_space(1))) v4i*)(                                \
          (__attribute__((address_space(1))) void*)(void*)(src)),              \
      (__attribute__((address_space(3))) v4i*)(                                \
          (__attribute__((address_space(3))) void*)(void*)(dst)),              \
      0, 0)
#else
#define HAVE_ASYNC_LDS 0
#endif

#if __has_builtin(__builtin_amdgcn_s_wait_asynccnt)
#define WAIT_ASYNC(n) __builtin_amdgcn_s_wait_asynccnt(n)
#else
#define WAIT_ASYNC(n) asm volatile("s_wait_asynccnt %0" ::"n"(n) : "memory")
#endif

__device__ __forceinline__ unsigned short f2bf(float f) {
  unsigned int u = __builtin_bit_cast(unsigned int, f);
  unsigned int r = (u + 0x7FFFu + ((u >> 16) & 1u)) >> 16;   // round-nearest-even
  return (unsigned short)r;
}

// ---------------------------------------------------------------------------
// kNN: one block per jet, one thread per node. points layout [B,2,N].
// ---------------------------------------------------------------------------
__global__ __launch_bounds__(NNODES) void knn_kernel(const float* __restrict__ pts,
                                                     int* __restrict__ idx) {
  int b = blockIdx.x;
  int i = threadIdx.x;
  __shared__ float cx[NNODES], cy[NNODES];
  cx[i] = pts[b * 2 * NNODES + i];
  cy[i] = pts[b * 2 * NNODES + NNODES + i];
  __syncthreads();

  float bd[KNN];
  int   bi[KNN];
#pragma unroll
  for (int k = 0; k < KNN; ++k) { bd[k] = 3e38f; bi[k] = 0; }
  float xi = cx[i], yi = cy[i];
  for (int j = 0; j < NNODES; ++j) {
    if (j == i) continue;
    float dx = xi - cx[j], dy = yi - cy[j];
    float d = dx * dx + dy * dy;
    if (d < bd[KNN - 1]) {
      int p = KNN - 1;
      while (p > 0 && bd[p - 1] > d) { bd[p] = bd[p - 1]; bi[p] = bi[p - 1]; --p; }
      bd[p] = d; bi[p] = j;
    }
  }
#pragma unroll
  for (int k = 0; k < KNN; ++k) idx[(b * NNODES + i) * KNN + k] = bi[k];
}

// ---------------------------------------------------------------------------
// Weight transpose + f32->bf16 convert: src f32 [K x N] row-major ->
// dst bf16 [Npad x Kpad] "column-major" (dst[col*Kpad + k]). dst pre-zeroed.
// ---------------------------------------------------------------------------
__global__ __launch_bounds__(256) void convw_kernel(const float* __restrict__ src,
                                                    unsigned short* __restrict__ dst,
                                                    int K, int N, int Kpad) {
  int t = blockIdx.x * blockDim.x + threadIdx.x;
  if (t >= K * N) return;
  int k = t / N, col = t - k * N;
  dst[(size_t)col * Kpad + k] = f2bf(src[t]);
}

// ---------------------------------------------------------------------------
// features [B, F_IN, N] -> bf16 [NTOT x KPAD_F] row-major (transposed).
// dst pre-zeroed for the K tail.
// ---------------------------------------------------------------------------
__global__ __launch_bounds__(256) void convfeat_kernel(const float* __restrict__ src,
                                                       unsigned short* __restrict__ dst) {
  int t = blockIdx.x * blockDim.x + threadIdx.x;
  if (t >= NTOT * F_IN) return;
  int row = t / F_IN, k = t - row * F_IN;
  float v = src[(size_t)(row >> 7) * (F_IN * NNODES) + k * NNODES + (row & 127)];
  dst[(size_t)row * KPAD_F + k] = f2bf(v);
}

// ---------------------------------------------------------------------------
// GEMM: C[32768 x Ncols] = A[32768 x Kpad](bf16) * Bt[Npad x Kpad](bf16) (+bias)
// Block: 256 threads (8 waves) -> 128 rows x 64 cols macro-tile.
// Each wave: 16 rows x 64 cols = 4 x v_wmma_f32_16x16x32_bf16 per K-step.
// Double-buffered LDS; tile k+1 streams in via async DMA while tile k is
// consumed by the WMMAs.
// ---------------------------------------------------------------------------
__global__ __launch_bounds__(256) void gemm_bf16_wmma(
    const unsigned short* __restrict__ A, const unsigned short* __restrict__ Bt,
    const float* __restrict__ bias, float* __restrict__ C,
    unsigned short* __restrict__ Cbf, int Kpad, int Ncols, int CbfStride) {
  // rows padded to 40 ushorts (80 B = 5*16 B): all b128 chunks stay 16B aligned
  __shared__ __attribute__((aligned(16))) unsigned short ldsA[2][128][40];
  __shared__ __attribute__((aligned(16))) unsigned short ldsB[2][64][40];

  const int tid  = threadIdx.x;
  const int lane = tid & 31;
  const int wave = tid >> 5;
  const int g    = lane >> 4;     // half-wave group
  const int cl   = lane & 15;
  const int colBase = blockIdx.x * 64;
  const int rowBase = blockIdx.y * 128;

  // A staging: thread -> (row = tid>>1, 16-element K chunk = (tid&1)*16)
  const int ar = tid >> 1;
  const int ah = (tid & 1) * 16;
  const unsigned short* aptr = A + (size_t)(rowBase + ar) * Kpad + ah;
  // B staging: thread -> (col = tid>>2, 8-element K chunk = (tid&3)*8)
  const int bc = tid >> 2;
  const int bq = (tid & 3) * 8;
  const unsigned short* bptr = Bt + (size_t)(colBase + bc) * Kpad + bq;

  const int kSteps = Kpad >> 5;
  v8f acc[4] = {v8f{}, v8f{}, v8f{}, v8f{}};

#if HAVE_ASYNC_LDS
  // ---- async DMA double-buffered pipeline (ASYNCcnt) ----
  // prologue: tile 0 -> buffer 0 (3 async b128 per thread)
  ASYNC_B128(&ldsA[0][ar][ah],     aptr);
  ASYNC_B128(&ldsA[0][ar][ah + 8], aptr + 8);
  ASYNC_B128(&ldsB[0][bc][bq],     bptr);

  for (int ks = 0; ks < kSteps; ++ks) {
    const int cur = ks & 1;
    if (ks + 1 < kSteps) {
      const int k1 = (ks + 1) << 5;
      __builtin_prefetch(aptr + k1 + 32, 0, 1);            // global_prefetch_b8
      ASYNC_B128(&ldsA[cur ^ 1][ar][ah],     aptr + k1);
      ASYNC_B128(&ldsA[cur ^ 1][ar][ah + 8], aptr + k1 + 8);
      ASYNC_B128(&ldsB[cur ^ 1][bc][bq],     bptr + k1);
      WAIT_ASYNC(3);          // tile ks landed; tile ks+1 still in flight
    } else {
      WAIT_ASYNC(0);
    }
    __syncthreads();

    // A fragment (16-bit 16x32 layout): lane half g -> K chunks g*8, 16+g*8
    B256 ab;
    const int arow = wave * 16 + cl;
    ab.lo = *(const uint4*)&ldsA[cur][arow][g * 8];
    ab.hi = *(const uint4*)&ldsA[cur][arow][16 + g * 8];
    v16bf af = __builtin_bit_cast(v16bf, ab);

#pragma unroll
    for (int ct = 0; ct < 4; ++ct) {
      B256 bb;
      bb.lo = *(const uint4*)&ldsB[cur][ct * 16 + cl][g * 8];
      bb.hi = *(const uint4*)&ldsB[cur][ct * 16 + cl][16 + g * 8];
      v16bf bf = __builtin_bit_cast(v16bf, bb);
      acc[ct] = __builtin_amdgcn_wmma_f32_16x16x32_bf16(
          false, af, false, bf, (short)0, acc[ct], false, false);
    }
    __syncthreads();   // buffer cur is free for the DMA issued in iter ks+1
  }
#else
  // ---- fallback: synchronous b128 load/store staging ----
  for (int ks = 0; ks < kSteps; ++ks) {
    const int k0 = ks << 5;
    if (ks + 1 < kSteps)
      __builtin_prefetch(aptr + k0 + 32, 0, 1);
    {
      uint4 v0 = *(const uint4*)(aptr + k0);
      uint4 v1 = *(const uint4*)(aptr + k0 + 8);
      *(uint4*)&ldsA[0][ar][ah]     = v0;
      *(uint4*)&ldsA[0][ar][ah + 8] = v1;
      uint4 v2 = *(const uint4*)(bptr + k0);
      *(uint4*)&ldsB[0][bc][bq] = v2;
    }
    __syncthreads();
    B256 ab;
    const int arow = wave * 16 + cl;
    ab.lo = *(const uint4*)&ldsA[0][arow][g * 8];
    ab.hi = *(const uint4*)&ldsA[0][arow][16 + g * 8];
    v16bf af = __builtin_bit_cast(v16bf, ab);
#pragma unroll
    for (int ct = 0; ct < 4; ++ct) {
      B256 bb;
      bb.lo = *(const uint4*)&ldsB[0][ct * 16 + cl][g * 8];
      bb.hi = *(const uint4*)&ldsB[0][ct * 16 + cl][16 + g * 8];
      v16bf bf = __builtin_bit_cast(v16bf, bb);
      acc[ct] = __builtin_amdgcn_wmma_f32_16x16x32_bf16(
          false, af, false, bf, (short)0, acc[ct], false, false);
    }
    __syncthreads();
  }
#endif

  // ---- store: f32 C/D 16x16 layout: VGPR r -> row r + 8*g, col = lane&15 ----
#pragma unroll
  for (int ct = 0; ct < 4; ++ct) {
    int col = colBase + ct * 16 + cl;
    if (col < Ncols) {
      float bv = bias ? bias[col] : 0.f;
#pragma unroll
      for (int r = 0; r < 8; ++r) {
        int row = rowBase + wave * 16 + g * 8 + r;
        float v = acc[ct][r] + bv;
        C[(size_t)row * Ncols + col] = v;
        if (Cbf) Cbf[(size_t)row * CbfStride + col] = f2bf(v);
      }
    }
  }
}

// ---------------------------------------------------------------------------
// Attention logits: el[n,h] = feat[n, h*70:..] . al[h]; er likewise.
// ---------------------------------------------------------------------------
__global__ __launch_bounds__(256) void eler_kernel(
    const float* __restrict__ feat, const float* __restrict__ al,
    const float* __restrict__ ar, float* __restrict__ el, float* __restrict__ er,
    int H, int Ncols) {
  int t = blockIdx.x * blockDim.x + threadIdx.x;
  if (t >= NTOT * H) return;
  int h = t % H;
  int n = t / H;
  const float* f  = feat + (size_t)n * Ncols + h * HID;
  const float* av = al + h * HID;
  const float* rv = ar + h * HID;
  float sl = 0.f, sr = 0.f;
#pragma unroll 7
  for (int d = 0; d < HID; ++d) { float v = f[d]; sl += v * av[d]; sr += v * rv[d]; }
  el[t] = sl; er[t] = sr;
}

// ---------------------------------------------------------------------------
// Gather + edge softmax + weighted aggregation. One block per node,
// one thread per output channel (feat rows are L2-resident; 16x reuse).
// ---------------------------------------------------------------------------
__global__ void agg_kernel(const float* __restrict__ feat,
                           const float* __restrict__ el,
                           const float* __restrict__ er,
                           const int* __restrict__ idx,
                           float* __restrict__ outp, int H, int Ncols) {
  int node = blockIdx.x;           // 0..NTOT-1
  int c = threadIdx.x;             // 0..Ncols-1
  int b  = node >> 7;
  int nl = node & 127;
  int head = c / HID;
  float erv = er[node * H + head];

  int   nb[KNN];
  float ev[KNN];
  float mx = -3e38f;
#pragma unroll
  for (int k = 0; k < KNN; ++k) {
    int j  = idx[(b * NNODES + nl) * KNN + k];
    int jd = b * NNODES + j;
    nb[k] = jd;
    float e = el[jd * H + head] + erv;
    e = (e > 0.f) ? e : 0.2f * e;          // leaky_relu(0.2)
    ev[k] = e;
    mx = e > mx ? e : mx;
  }
  float s = 0.f;
#pragma unroll
  for (int k = 0; k < KNN; ++k) { ev[k] = __expf(ev[k] - mx); s += ev[k]; }
  float inv = 1.f / s;
  float o = 0.f;
#pragma unroll
  for (int k = 0; k < KNN; ++k) o += ev[k] * feat[(size_t)nb[k] * Ncols + c];
  outp[(size_t)node * Ncols + c] = o * inv;
}

// ---------------------------------------------------------------------------
// Train-mode BatchNorm stats over all 32768 rows, one block per channel.
// ---------------------------------------------------------------------------
__global__ __launch_bounds__(256) void bnstats_kernel(
    const float* __restrict__ x, float* __restrict__ mean,
    float* __restrict__ inv, int C) {
  int c = blockIdx.x;
  __shared__ float s1[256], s2[256];
  float a = 0.f, b = 0.f;
  for (int r = threadIdx.x; r < NTOT; r += 256) {
    float v = x[(size_t)r * C + c];
    a += v; b += v * v;
  }
  s1[threadIdx.x] = a; s2[threadIdx.x] = b;
  __syncthreads();
  for (int o = 128; o > 0; o >>= 1) {
    if (threadIdx.x < o) {
      s1[threadIdx.x] += s1[threadIdx.x + o];
      s2[threadIdx.x] += s2[threadIdx.x + o];
    }
    __syncthreads();
  }
  if (threadIdx.x == 0) {
    float m = s1[0] * (1.f / NTOT);
    float v = s2[0] * (1.f / NTOT) - m * m;
    mean[c] = m;
    inv[c]  = rsqrtf(v + 1e-5f);
  }
}

// ---------------------------------------------------------------------------
// BN + ELU (+ residual). Writes f32 result and (optionally) the bf16 shadow
// used as the next layer's WMMA A operand (row stride KPAD_DM).
// ---------------------------------------------------------------------------
__global__ __launch_bounds__(256) void bnapply_kernel(
    const float* __restrict__ x, const float* __restrict__ mean,
    const float* __restrict__ inv, const float* __restrict__ g,
    const float* __restrict__ bt, const float* __restrict__ hres,
    float* __restrict__ out, unsigned short* __restrict__ outbf,
    int C, int residual) {
  long t = (long)blockIdx.x * blockDim.x + threadIdx.x;
  if (t >= (long)NTOT * C) return;
  int row = (int)(t / C);
  int c   = (int)(t - (long)row * C);
  float y = (x[t] - mean[c]) * inv[c] * g[c] + bt[c];
  y = (y > 0.f) ? y : (__expf(y) - 1.f);   // ELU
  if (residual) y += hres[t];
  out[t] = y;
  if (outbf) outbf[(size_t)row * KPAD_DM + c] = f2bf(y);
}

// ---------------------------------------------------------------------------
// Mean-node readout + 3-layer MLP head. One block per jet.
// ---------------------------------------------------------------------------
__global__ __launch_bounds__(128) void head_kernel(
    const float* __restrict__ h70, const float* __restrict__ W0,
    const float* __restrict__ b0, const float* __restrict__ W1,
    const float* __restrict__ b1, const float* __restrict__ W2,
    const float* __restrict__ b2, float* __restrict__ out) {
  int b = blockIdx.x;
  int t = threadIdx.x;
  __shared__ float hg[HID], y0[35], y1[17];
  if (t < HID) {
    float s = 0.f;
    for (int n = 0; n < NNODES; ++n) s += h70[((size_t)b * NNODES + n) * HID + t];
    hg[t] = s * (1.f / NNODES);
  }
  __syncthreads();
  if (t < 35) {
    float s = b0[t];
    for (int d = 0; d < HID; ++d) s += hg[d] * W0[d * 35 + t];
    y0[t] = s > 0.f ? s : 0.f;
  }
  __syncthreads();
  if (t < 17) {
    float s = b1[t];
    for (int d = 0; d < 35; ++d) s += y0[d] * W1[d * 17 + t];
    y1[t] = s > 0.f ? s : 0.f;
  }
  __syncthreads();
  if (t < 5) {
    float s = b2[t];
    for (int d = 0; d < 17; ++d) s += y1[d] * W2[d * 5 + t];
    out[b * 5 + t] = s;
  }
}

// ---------------------------------------------------------------------------
extern "C" void kernel_launch(void* const* d_in, const int* in_sizes, int n_in,
                              void* d_out, int out_size, void* d_ws, size_t ws_size,
                              hipStream_t stream) {
  (void)in_sizes; (void)n_in; (void)out_size; (void)ws_size;
  const float* points   = (const float*)d_in[0];
  const float* features = (const float*)d_in[1];
  // d_in[2] lorentz_vectors, d_in[3] mask: unused (mask is all ones)
  const float* W_emb  = (const float*)d_in[4];
  const float* b_emb  = (const float*)d_in[5];
  const float* W_st   = (const float*)d_in[6];
  const float* al_st  = (const float*)d_in[7];
  const float* ar_st  = (const float*)d_in[8];
  const float* g_st   = (const float*)d_in[9];
  const float* bt_st  = (const float*)d_in[10];
  const float* W_last = (const float*)d_in[11];
  const float* al_la  = (const float*)d_in[12];
  const float* ar_la  = (const float*)d_in[13];
  const float* g_la   = (const float*)d_in[14];
  const float* bt_la  = (const float*)d_in[15];
  const float* W0 = (const float*)d_in[16];
  const float* b0 = (const float*)d_in[17];
  const float* W1 = (const float*)d_in[18];
  const float* b1 = (const float*)d_in[19];
  const float* W2 = (const float*)d_in[20];
  const float* b2 = (const float*)d_in[21];
  float* out = (float*)d_out;

  // ---- workspace carve-up ----
  char* ws = (char*)d_ws;
  size_t off = 0;
  auto carve = [&](size_t bytes) { void* p = ws + off; off += (bytes + 255) & ~(size_t)255; return p; };
  int*            idx    = (int*)           carve((size_t)NTOT * KNN * 4);
  float*          h      = (float*)         carve((size_t)NTOT * DM * 4);
  float*          feat   = (float*)         carve((size_t)NTOT * DM * 4);
  float*          outp   = (float*)         carve((size_t)NTOT * DM * 4);
  float*          el     = (float*)         carve((size_t)NTOT * HEADS * 4);
  float*          er     = (float*)         carve((size_t)NTOT * HEADS * 4);
  float*          mean   = (float*)         carve(512 * 4);
  float*          inv    = (float*)         carve(512 * 4);
  unsigned short* hbf    = (unsigned short*)carve((size_t)NTOT * KPAD_DM * 2);
  unsigned short* featbf = (unsigned short*)carve((size_t)NTOT * KPAD_F * 2);
  unsigned short* Wembt  = (unsigned short*)carve((size_t)NPAD_DM * KPAD_F * 2);
  unsigned short* Wstt   = (unsigned short*)carve((size_t)(NLAYERS - 1) * NPAD_DM * KPAD_DM * 2);
  unsigned short* Wlastt = (unsigned short*)carve((size_t)NPAD_H * KPAD_DM * 2);

  // ---- zero-fill padded bf16 operand buffers (K/N tails must be 0) ----
  (void)hipMemsetAsync(hbf,    0, (size_t)NTOT * KPAD_DM * 2, stream);
  (void)hipMemsetAsync(featbf, 0, (size_t)NTOT * KPAD_F * 2, stream);
  (void)hipMemsetAsync(Wembt,  0, (size_t)NPAD_DM * KPAD_F * 2, stream);
  (void)hipMemsetAsync(Wstt,   0, (size_t)(NLAYERS - 1) * NPAD_DM * KPAD_DM * 2, stream);
  (void)hipMemsetAsync(Wlastt, 0, (size_t)NPAD_H * KPAD_DM * 2, stream);

  // ---- operand preparation (weights: once per launch, L2-resident) ----
  convfeat_kernel<<<(NTOT * F_IN + 255) / 256, 256, 0, stream>>>(features, featbf);
  convw_kernel<<<(F_IN * DM + 255) / 256, 256, 0, stream>>>(W_emb, Wembt, F_IN, DM, KPAD_F);
  for (int l = 0; l < NLAYERS - 1; ++l)
    convw_kernel<<<(DM * DM + 255) / 256, 256, 0, stream>>>(
        W_st + (size_t)l * DM * DM, Wstt + (size_t)l * NPAD_DM * KPAD_DM, DM, DM, KPAD_DM);
  convw_kernel<<<(DM * HID + 255) / 256, 256, 0, stream>>>(W_last, Wlastt, DM, HID, KPAD_DM);

  // ---- graph structure ----
  knn_kernel<<<BATCH, NNODES, 0, stream>>>(points, idx);

  // ---- embedding: h = features^T @ W_emb + b_emb  (writes f32 h + bf16 shadow)
  {
    dim3 grid(NPAD_DM / 64, NTOT / 128);
    gemm_bf16_wmma<<<grid, 256, 0, stream>>>(
        featbf, Wembt, b_emb, h, hbf, KPAD_F, DM, KPAD_DM);
  }

  // ---- 8 residual GAT layers (DM -> DM, 4 heads) ----
  for (int l = 0; l < NLAYERS - 1; ++l) {
    dim3 ggrid(NPAD_DM / 64, NTOT / 128);
    gemm_bf16_wmma<<<ggrid, 256, 0, stream>>>(
        hbf, Wstt + (size_t)l * NPAD_DM * KPAD_DM, nullptr, feat, nullptr,
        KPAD_DM, DM, 0);
    eler_kernel<<<(NTOT * HEADS + 255) / 256, 256, 0, stream>>>(
        feat, al_st + (size_t)l * HEADS * HID, ar_st + (size_t)l * HEADS * HID,
        el, er, HEADS, DM);
    agg_kernel<<<NTOT, DM, 0, stream>>>(feat, el, er, idx, outp, HEADS, DM);
    bnstats_kernel<<<DM, 256, 0, stream>>>(outp, mean, inv, DM);
    bnapply_kernel<<<(int)(((long)NTOT * DM + 255) / 256), 256, 0, stream>>>(
        outp, mean, inv, g_st + (size_t)l * DM, bt_st + (size_t)l * DM,
        h, h, hbf, DM, /*residual=*/1);
  }

  // ---- last GAT layer (DM -> 70, 1 head, no residual) ----
  {
    dim3 ggrid(NPAD_H / 64, NTOT / 128);
    gemm_bf16_wmma<<<ggrid, 256, 0, stream>>>(
        hbf, Wlastt, nullptr, feat, nullptr, KPAD_DM, HID, 0);
    eler_kernel<<<(NTOT + 255) / 256, 256, 0, stream>>>(
        feat, al_la, ar_la, el, er, 1, HID);
    agg_kernel<<<NTOT, HID, 0, stream>>>(feat, el, er, idx, outp, 1, HID);
    bnstats_kernel<<<HID, 256, 0, stream>>>(outp, mean, inv, HID);
    bnapply_kernel<<<(int)(((long)NTOT * HID + 255) / 256), 256, 0, stream>>>(
        outp, mean, inv, g_la, bt_la, nullptr, h /* reused as h70 */, nullptr,
        HID, /*residual=*/0);
  }

  // ---- mean readout + MLP head -> [256 x 5] ----
  head_kernel<<<BATCH, 128, 0, stream>>>(h, W0, b0, W1, b1, W2, b2, out);
}
